// HV_attention_20804821582391
// MI455X (gfx1250) — compile-verified, matrix operand order checked by reading, not run on hardware
//
#include <hip/hip_runtime.h>

typedef __attribute__((ext_vector_type(16))) _Float16 v16h;
typedef __attribute__((ext_vector_type(8)))  float    v8f;

// ---------------- WMMA helpers (gfx1250, wave32) ----------------

__device__ __forceinline__ v8f wmma32(v16h a, v16h b, v8f c) {
  return __builtin_amdgcn_wmma_f32_16x16x32_f16(false, a, false, b, (short)0, c, false, false);
}

// Load a 16x32 f16 fragment from a row-major matrix (LDS or generic), row stride `ld` halves.
// Lane l holds row (l&15); VGPR v holds K pair per ISA 16-bit A-matrix 16x32 layout.
__device__ __forceinline__ v16h frag_ld(const _Float16* p, int ld) {
  const int lane = threadIdx.x & 31;
  const _Float16* r = p + (lane & 15) * ld;
  const int hb = (lane >> 4) << 3;   // 0 or 8
  v16h f;
#pragma unroll
  for (int v = 0; v < 4; ++v) {
    f[2*v]     = r[hb + 2*v];
    f[2*v + 1] = r[hb + 2*v + 1];
    f[8 + 2*v]     = r[16 + hb + 2*v];
    f[8 + 2*v + 1] = r[16 + hb + 2*v + 1];
  }
  return f;
}

// Same, global pointer with 64-bit row stride.
__device__ __forceinline__ v16h frag_ldg(const _Float16* __restrict__ p, size_t ld) {
  const int lane = threadIdx.x & 31;
  const _Float16* __restrict__ r = p + (size_t)(lane & 15) * ld;
  const int hb = (lane >> 4) << 3;
  v16h f;
#pragma unroll
  for (int v = 0; v < 4; ++v) {
    f[2*v]     = r[hb + 2*v];
    f[2*v + 1] = r[hb + 2*v + 1];
    f[8 + 2*v]     = r[16 + hb + 2*v];
    f[8 + 2*v + 1] = r[16 + hb + 2*v + 1];
  }
  return f;
}

// ---------------- generic f16 WMMA GEMM:  out[M,N] = A[M,K] * W[N,K]^T + bias ----------------
// BM=128 BN=64 BK=32, 8 waves, each wave does a 32x32 (2x2 WMMA tiles).
// remap=1: output rows are vertical-attention rows r2 -> remapped to (h,w,b) row order.

__global__ __launch_bounds__(256) void gemm_f16(
    const _Float16* __restrict__ A, const _Float16* __restrict__ W,
    const float* __restrict__ bias, float* __restrict__ outF,
    _Float16* __restrict__ outH, int M, int N, int K, int remap)
{
  __shared__ _Float16 As[128][40];
  __shared__ _Float16 Wsh[64][40];
  const int m0 = blockIdx.x * 128;
  const int n0 = blockIdx.y * 64;
  const int tid = threadIdx.x;
  const int wid = tid >> 5, lane = tid & 31;
  const int wm = wid & 3, wn = wid >> 2;

  v8f acc00 = {}, acc01 = {}, acc10 = {}, acc11 = {};

  for (int k0 = 0; k0 < K; k0 += 32) {
#pragma unroll
    for (int i = 0; i < 2; ++i) {               // 128x32 A tile, 16B vector loads
      int id = tid + (i << 8);
      int r = id >> 2, cc = (id & 3) << 3;
      *(uint4*)&As[r][cc] = *(const uint4*)(A + (size_t)(m0 + r) * K + k0 + cc);
    }
    {                                            // 64x32 W tile (guard rows vs N)
      int r = tid >> 2, cc = (tid & 3) << 3;
      uint4 v = {0u, 0u, 0u, 0u};
      if (n0 + r < N) v = *(const uint4*)(W + (size_t)(n0 + r) * K + k0 + cc);
      *(uint4*)&Wsh[r][cc] = v;
    }
    __syncthreads();
    v16h a0 = frag_ld(&As[wm * 32][0], 40);
    v16h a1 = frag_ld(&As[wm * 32 + 16][0], 40);
    v16h b0 = frag_ld(&Wsh[wn * 32][0], 40);
    v16h b1 = frag_ld(&Wsh[wn * 32 + 16][0], 40);
    acc00 = wmma32(a0, b0, acc00);
    acc01 = wmma32(a0, b1, acc01);
    acc10 = wmma32(a1, b0, acc10);
    acc11 = wmma32(a1, b1, acc11);
    __syncthreads();
  }

  const int colo = lane & 15, halfo = (lane >> 4) << 3;
  auto store_tile = [&](v8f acc, int mt, int nt) {
    int n = n0 + wn * 32 + nt * 16 + colo;
    if (n >= N) return;
    float bv = bias ? bias[n] : 0.0f;
#pragma unroll
    for (int v = 0; v < 8; ++v) {
      int m = m0 + wm * 32 + mt * 16 + halfo + v;
      float val = acc[v] + bv;
      if (outH) outH[(size_t)m * N + n] = (_Float16)val;
      if (outF) {
        size_t orow = (size_t)m;
        if (remap) {                 // m = r2 = (hh*4+wd)*128 + bb*16+nwi -> (hh*64+nwi*4+wd)*8+bb
          int t2 = m >> 7, b2 = m & 127;
          int hh = t2 >> 2, wd = t2 & 3;
          int bb = b2 >> 4, nwi = b2 & 15;
          orow = (size_t)((hh << 6) + (nwi << 2) + wd) * 8 + bb;
        }
        outF[orow * (size_t)N + n] = val;
      }
    }
  };
  store_tile(acc00, 0, 0); store_tile(acc01, 0, 1);
  store_tile(acc10, 1, 0); store_tile(acc11, 1, 1);
}

// ---------------- attention: per (batch,head), 32-query block, Lk=256, D=64 ----------------
// QP/KP rows are (token*Brows + bi), cols head*64+k.  VT is [bh][64 d][256 n] f16.

__global__ __launch_bounds__(256) void attn_kernel(
    const _Float16* __restrict__ QP, const _Float16* __restrict__ KP,
    const _Float16* __restrict__ VT, _Float16* __restrict__ O,
    int Brows, float scale)
{
  __shared__ float    S[32][256];
  __shared__ _Float16 P[32][264];
  __shared__ float    rsum[32];

  const int qb   = blockIdx.x * 32;
  const int bh   = blockIdx.y;
  const int bi   = bh >> 3;
  const int head = bh & 7;
  const int tid = threadIdx.x, wid = tid >> 5, lane = tid & 31;
  const size_t rstr = (size_t)Brows * 512;
  const int chb = head * 64;

  // ---- S = Q * K^T (2 m-tiles x 16 n-tiles, 4 tiles per wave) ----
  for (int t = wid; t < 32; t += 8) {
    int mt = t >> 4, nt = t & 15;
    v8f acc = {};
#pragma unroll
    for (int ks = 0; ks < 64; ks += 32) {
      v16h a = frag_ldg(QP + ((size_t)(qb + mt * 16) * Brows + bi) * 512 + chb + ks, rstr);
      v16h b = frag_ldg(KP + ((size_t)(nt * 16) * Brows + bi) * 512 + chb + ks, rstr);
      acc = wmma32(a, b, acc);
    }
    int col = nt * 16 + (lane & 15);
    int mrow = mt * 16 + ((lane >> 4) << 3);
#pragma unroll
    for (int v = 0; v < 8; ++v) S[mrow + v][col] = acc[v];
  }
  __syncthreads();

  // ---- softmax rows (4 rows per wave) ----
  for (int r = wid * 4; r < wid * 4 + 4; ++r) {
    float mx = -1e30f;
    for (int c = lane; c < 256; c += 32) mx = fmaxf(mx, S[r][c] * scale);
#pragma unroll
    for (int o = 16; o > 0; o >>= 1) mx = fmaxf(mx, __shfl_xor(mx, o, 32));
    float sum = 0.0f;
    for (int c = lane; c < 256; c += 32) {
      float e = __expf(S[r][c] * scale - mx);
      P[r][c] = (_Float16)e;
      sum += e;
    }
#pragma unroll
    for (int o = 16; o > 0; o >>= 1) sum += __shfl_xor(sum, o, 32);
    if (lane == 0) rsum[r] = sum;
  }
  __syncthreads();

  // ---- O = P * V  (2 m-tiles x 4 d-tiles = 8 tiles, 1 per wave) ----
  {
    int mt = wid >> 2, nt = wid & 3;
    v8f acc = {};
#pragma unroll
    for (int ks = 0; ks < 256; ks += 32) {
      v16h a = frag_ld(&P[mt * 16][ks], 264);
      v16h b = frag_ldg(VT + ((size_t)bh * 64 + nt * 16) * 256 + ks, 256);
      acc = wmma32(a, b, acc);
    }
    int col = nt * 16 + (lane & 15);
    int mbase = mt * 16 + ((lane >> 4) << 3);
#pragma unroll
    for (int v = 0; v < 8; ++v) {
      int m = mbase + v;
      float val = acc[v] / rsum[m];
      O[((size_t)(qb + m) * Brows + bi) * 512 + chb + col] = (_Float16)val;
    }
  }
}

// ---------------- data movement / pack kernels ----------------

__global__ void f32_to_f16_k(const float* __restrict__ s, _Float16* __restrict__ d, size_t n) {
  size_t i = (size_t)blockIdx.x * 256 + threadIdx.x;
  if (i < n) d[i] = (_Float16)s[i];
}

// Q tokens: Q[(t*8+b)*512+c] = x[b][c][t]
__global__ void pack_q_k(const float* __restrict__ x, _Float16* __restrict__ Q) {
  size_t idx = (size_t)blockIdx.x * 256 + threadIdx.x;    // 16,777,216
  int t = (int)(idx & 4095);
  int c = (int)((idx >> 12) & 511);
  int b = (int)(idx >> 21);
  Q[((size_t)t * 8 + b) * 512 + c] = (_Float16)x[idx];
}

// pooled KV tokens: KV[(n*8+b)*512+c] = mean 4x4
__global__ void pack_pool_k(const float* __restrict__ x, _Float16* __restrict__ KV) {
  size_t idx = (size_t)blockIdx.x * 256 + threadIdx.x;    // 1,048,576  (b,c,n)
  int n = (int)(idx & 255);
  int c = (int)((idx >> 8) & 511);
  int b = (int)(idx >> 17);
  int ph = n >> 4, pw = n & 15;
  const float* base = x + (((size_t)b * 512 + c) << 12) + (ph << 2) * 64 + (pw << 2);
  float s = 0.0f;
#pragma unroll
  for (int i = 0; i < 4; ++i)
#pragma unroll
    for (int j = 0; j < 4; ++j) s += base[i * 64 + j];
  KV[((size_t)n * 8 + b) * 512 + c] = (_Float16)(s * 0.0625f);
}

// vertical tokens: Xv[r2*512+c] = x[b][c][hh][nwi*4+wd], r2=(hh*4+wd)*128+b*16+nwi
__global__ void pack_xv_k(const float* __restrict__ x, _Float16* __restrict__ Xv) {
  size_t idx = (size_t)blockIdx.x * 256 + threadIdx.x;    // 16,777,216
  int t = (int)(idx & 4095);
  int c = (int)((idx >> 12) & 511);
  int b = (int)(idx >> 21);
  int hh = t >> 6, ww = t & 63;
  int wd = ww & 3, nwi = ww >> 2;
  int r2 = ((hh << 2) + wd) * 128 + (b << 4) + nwi;
  Xv[(size_t)r2 * 512 + c] = (_Float16)x[idx];
}

// VP[(n<<bshift)+b][head*64+d] -> VT[((b*8+head)*64+d)*256+n]
__global__ void transpose_vt_k(const _Float16* __restrict__ VP, _Float16* __restrict__ VT,
                               int bshift, size_t total) {
  size_t idx = (size_t)blockIdx.x * 256 + threadIdx.x;
  if (idx >= total) return;
  int c = (int)(idx & 511);
  size_t row = idx >> 9;
  int n = (int)(row >> bshift);
  int b = (int)(row & (((size_t)1 << bshift) - 1));
  int head = c >> 6, d = c & 63;
  VT[(((size_t)b * 8 + head) * 64 + d) * 256 + n] = VP[idx];
}

// ---------------- concat + residual + partial LN stats (deterministic) ----------------

__global__ __launch_bounds__(256) void assemble_k(
    const float* __restrict__ x, const float* __restrict__ HA, const float* __restrict__ VA,
    float* __restrict__ out4, float* __restrict__ res,
    float* __restrict__ part1, float* __restrict__ part2)
{
  size_t idx = (size_t)blockIdx.x * 256 + threadIdx.x;    // (b, ch, p) over 16,777,216
  int p  = (int)(idx & 4095);
  int ch = (int)((idx >> 12) & 511);
  int bb = (int)(idx >> 21);
  int hh = p >> 6, ww = p & 63;
  size_t rowq = (size_t)p * 8 + bb;
  float a;
  if (ch < 358) {
    a = HA[rowq * 358 + ch];
  } else {
    int r2 = ((hh << 2) + (ww & 3)) * 128 + (bb << 4) + (ww >> 2);
    a = VA[(size_t)r2 * 154 + (ch - 358)];
  }
  out4[idx] = a;
  float r = a + x[idx];
  res[idx] = r;

  float s1 = r, s2 = r * r;
#pragma unroll
  for (int o = 16; o > 0; o >>= 1) { s1 += __shfl_xor(s1, o, 32); s2 += __shfl_xor(s2, o, 32); }
  __shared__ float w1[8], w2[8];
  int wid = threadIdx.x >> 5, lane = threadIdx.x & 31;
  if (lane == 0) { w1[wid] = s1; w2[wid] = s2; }
  __syncthreads();
  if (threadIdx.x == 0) {
    float t1 = 0.0f, t2 = 0.0f;
#pragma unroll
    for (int k = 0; k < 8; ++k) { t1 += w1[k]; t2 += w2[k]; }
    part1[blockIdx.x] = t1;
    part2[blockIdx.x] = t2;
  }
}

__global__ __launch_bounds__(256) void reduce_sums_k(
    const float* __restrict__ p1, const float* __restrict__ p2, float* __restrict__ sums)
{
  int b = blockIdx.x, tid = threadIdx.x;
  float s1 = 0.0f, s2 = 0.0f;
  for (int i = tid; i < 8192; i += 256) { s1 += p1[b * 8192 + i]; s2 += p2[b * 8192 + i]; }
#pragma unroll
  for (int o = 16; o > 0; o >>= 1) { s1 += __shfl_xor(s1, o, 32); s2 += __shfl_xor(s2, o, 32); }
  __shared__ float w1[8], w2[8];
  int wid = tid >> 5, lane = tid & 31;
  if (lane == 0) { w1[wid] = s1; w2[wid] = s2; }
  __syncthreads();
  if (tid == 0) {
    float t1 = 0.0f, t2 = 0.0f;
#pragma unroll
    for (int k = 0; k < 8; ++k) { t1 += w1[k]; t2 += w2[k]; }
    sums[b] = t1; sums[8 + b] = t2;
  }
}

__global__ void finalize_k(float* __restrict__ out1, const float* __restrict__ sums) {
  size_t idx = (size_t)blockIdx.x * 256 + threadIdx.x;
  int bb = (int)(idx >> 21);
  const float inv = 1.0f / 2097152.0f;
  float mu = sums[bb] * inv;
  float var = sums[8 + bb] * inv - mu * mu;
  out1[idx] = (out1[idx] - mu) * rsqrtf(var + 1e-5f);
}

// ---------------- host side ----------------

extern "C" void kernel_launch(void* const* d_in, const int* in_sizes, int n_in,
                              void* d_out, int out_size, void* d_ws, size_t ws_size,
                              hipStream_t stream)
{
  (void)in_sizes; (void)n_in; (void)out_size; (void)ws_size;
  const float* x       = (const float*)d_in[0];
  const float* h_in_w  = (const float*)d_in[1];
  const float* h_in_b  = (const float*)d_in[2];
  const float* h_out_w = (const float*)d_in[3];
  const float* h_out_b = (const float*)d_in[4];
  const float* v_in_w  = (const float*)d_in[5];
  const float* v_in_b  = (const float*)d_in[6];
  const float* v_out_w = (const float*)d_in[7];
  const float* v_out_b = (const float*)d_in[8];
  const float* hfc_w   = (const float*)d_in[9];
  const float* hfc_b   = (const float*)d_in[10];
  const float* vfc_w   = (const float*)d_in[11];
  const float* vfc_b   = (const float*)d_in[12];

  float* out  = (float*)d_out;
  float* out1 = out;                       // result (res staged here, normalized in place)
  float* out2 = out + 16777216ull;         // ha_x  (h,w,b,c) == row-major (t*8+b, c)
  float* out3 = out + 2ull * 16777216ull;  // va_x  (written via remap store)
  float* out4 = out + 3ull * 16777216ull;  // attn_bchw

  char* ws = (char*)d_ws;
  size_t off = 0;
  auto alloc = [&](size_t bytes) -> char* {
    char* p = ws + off;
    off = (off + bytes + 255) & ~(size_t)255;
    return p;
  };

  // f16 weights
  _Float16* wHin  = (_Float16*)alloc(1536ull * 512 * 2);
  _Float16* wHout = (_Float16*)alloc(512ull * 512 * 2);
  _Float16* wVin  = (_Float16*)alloc(1536ull * 512 * 2);
  _Float16* wVout = (_Float16*)alloc(512ull * 512 * 2);
  _Float16* wHfc  = (_Float16*)alloc(358ull * 512 * 2);
  _Float16* wVfc  = (_Float16*)alloc(154ull * 512 * 2);
  // activations (big buffers reused across H->V phases)
  _Float16* Qtok = (_Float16*)alloc(32768ull * 512 * 2);  // later: Xv
  _Float16* KV   = (_Float16*)alloc(2048ull * 512 * 2);
  _Float16* QP   = (_Float16*)alloc(32768ull * 512 * 2);  // QPh, later QPv
  _Float16* KPh  = (_Float16*)alloc(2048ull * 512 * 2);
  _Float16* VPh  = (_Float16*)alloc(2048ull * 512 * 2);
  _Float16* VTh  = (_Float16*)alloc(8ull * 8 * 64 * 256 * 2);
  _Float16* OH   = (_Float16*)alloc(32768ull * 512 * 2);  // later KPv
  _Float16* HX16 = (_Float16*)alloc(32768ull * 512 * 2);  // later VPv
  _Float16* VTv  = (_Float16*)alloc(128ull * 8 * 64 * 256 * 2);
  _Float16* OV   = (_Float16*)alloc(32768ull * 512 * 2);
  _Float16* VO16 = (_Float16*)alloc(32768ull * 512 * 2);
  float*    HA   = (float*)alloc(32768ull * 358 * 4);
  float*    VA   = (float*)alloc(32768ull * 154 * 4);
  float*    part1 = (float*)alloc(65536ull * 4);
  float*    part2 = (float*)alloc(65536ull * 4);
  float*    sums  = (float*)alloc(16ull * 4);

  _Float16* Xv  = Qtok;
  _Float16* QPv = QP;
  _Float16* KPv = OH;
  _Float16* VPv = HX16;

  auto cvt = [&](const float* s, _Float16* d, size_t n) {
    f32_to_f16_k<<<(unsigned)((n + 255) / 256), 256, 0, stream>>>(s, d, n);
  };
  auto gemm = [&](const _Float16* A, const _Float16* W, const float* bias,
                  float* oF, _Float16* oH, int M, int N, int remap) {
    dim3 g(M / 128, (N + 63) / 64);
    gemm_f16<<<g, 256, 0, stream>>>(A, W, bias, oF, oH, M, N, 512, remap);
  };

  // weights -> f16
  cvt(h_in_w,  wHin,  1536ull * 512);
  cvt(h_out_w, wHout, 512ull * 512);
  cvt(v_in_w,  wVin,  1536ull * 512);
  cvt(v_out_w, wVout, 512ull * 512);
  cvt(hfc_w,   wHfc,  358ull * 512);
  cvt(vfc_w,   wVfc,  154ull * 512);

  // ---- horizontal branch ----
  pack_q_k<<<65536, 256, 0, stream>>>(x, Qtok);
  pack_pool_k<<<4096, 256, 0, stream>>>(x, KV);
  gemm(Qtok, wHin,              h_in_b,        nullptr, QP,  32768, 512, 0);
  gemm(KV,   wHin + 512 * 512,  h_in_b + 512,  nullptr, KPh,  2048, 512, 0);
  gemm(KV,   wHin + 1024 * 512, h_in_b + 1024, nullptr, VPh,  2048, 512, 0);
  transpose_vt_k<<<4096, 256, 0, stream>>>(VPh, VTh, 3, 2048ull * 512);
  attn_kernel<<<dim3(128, 64), 256, 0, stream>>>(QP, KPh, VTh, OH, 8, 0.125f);
  gemm(OH, wHout, h_out_b, out2, HX16, 32768, 512, 0);       // ha_x (f32 output) + f16 copy
  gemm(HX16, wHfc, hfc_b, HA, nullptr, 32768, 358, 0);       // ha

  // ---- vertical branch ----
  pack_xv_k<<<65536, 256, 0, stream>>>(x, Xv);
  gemm(Xv, wVin,              v_in_b,        nullptr, QPv, 32768, 512, 0);
  gemm(Xv, wVin + 512 * 512,  v_in_b + 512,  nullptr, KPv, 32768, 512, 0);
  gemm(Xv, wVin + 1024 * 512, v_in_b + 1024, nullptr, VPv, 32768, 512, 0);
  transpose_vt_k<<<65536, 256, 0, stream>>>(VPv, VTv, 7, 32768ull * 512);
  attn_kernel<<<dim3(8, 1024), 256, 0, stream>>>(QPv, KPv, VTv, OV, 128, 0.125f);
  gemm(OV, wVout, v_out_b, out3, VO16, 32768, 512, 1);       // va_x remapped to (h,w,b,c)
  gemm(VO16, wVfc, vfc_b, VA, nullptr, 32768, 154, 0);       // va

  // ---- concat + residual + layernorm ----
  assemble_k<<<65536, 256, 0, stream>>>(x, HA, VA, out4, out1, part1, part2);
  reduce_sums_k<<<8, 256, 0, stream>>>(part1, part2, sums);
  finalize_k<<<65536, 256, 0, stream>>>(out1, sums);
}
